// MCAMamba_36601711296730
// MI455X (gfx1250) — matrix operand order repaired and checked by reading, not compiled, not run on hardware
//
#include <hip/hip_runtime.h>
#include <hip/hip_bf16.h>

#define DEVINL __device__ __forceinline__

typedef __attribute__((ext_vector_type(16))) __bf16 bf16x16;
typedef __attribute__((ext_vector_type(8)))  float  f32x8;
typedef __attribute__((ext_vector_type(4)))  unsigned int u32x4;
typedef __attribute__((ext_vector_type(8)))  unsigned int u32x8;

// ---------------- problem constants ----------------
constexpr int kNHist = 8192;
constexpr int kMp    = 284;
constexpr int kN     = kNHist + kMp;     // 8476
constexpr int kDim   = 256;
constexpr int kHeads = 8;
constexpr int kDHead = 256;
constexpr int kInner = kHeads * kDHead;  // 2048
constexpr int kQKV   = 3 * kInner;       // 6144
constexpr int kKRes  = 33;
constexpr int kPadC  = 16;
constexpr float kEps = 1e-5f;

// padded extents (block-tile multiples) so tile loads need no guards
constexpr int kNp    = 8512;             // 8476 -> 133*64
constexpr int kMpPad = 320;              // 284  -> 5*64
constexpr int kKpPad = 288;              // 284  -> 9*32 (K pad, zero-filled)

// ---------------- GEMM tile config -----------------
constexpr int BM = 64, BN = 128, BK = 32, NTHR = 256;

DEVINL __bf16 f2bf(float f) {
  __hip_bfloat16 h = __float2bfloat16(f);
  return __builtin_bit_cast(__bf16, h);
}
DEVINL bf16x16 cat16(u32x4 lo, u32x4 hi) {
  u32x8 t = __builtin_shufflevector(lo, hi, 0, 1, 2, 3, 4, 5, 6, 7);
  return __builtin_bit_cast(bf16x16, t);
}

// ---------------- LayerNorm -> bf16 ----------------
__global__ __launch_bounds__(256) void ln_kernel(
    const float* __restrict__ x, const float* __restrict__ gamma,
    const float* __restrict__ beta, __bf16* __restrict__ xn) {
  const int n = blockIdx.x, d = threadIdx.x;
  float v = x[(size_t)n * kDim + d];
  float s = v, s2 = v * v;
  #pragma unroll
  for (int o = 16; o; o >>= 1) {
    s  += __shfl_xor(s,  o, 32);
    s2 += __shfl_xor(s2, o, 32);
  }
  __shared__ float sh[8], sh2[8];
  const int lane = threadIdx.x & 31, w = threadIdx.x >> 5;
  if (!lane) { sh[w] = s; sh2[w] = s2; }
  __syncthreads();
  float ts = 0.f, ts2 = 0.f;
  #pragma unroll
  for (int i = 0; i < 8; ++i) { ts += sh[i]; ts2 += sh2[i]; }
  const float mu  = ts * (1.f / kDim);
  const float var = ts2 * (1.f / kDim) - mu * mu;
  const float r   = rsqrtf(var + kEps);
  xn[(size_t)n * kDim + d] = f2bf((v - mu) * r * gamma[d] + beta[d]);
}

// ---------------- f32 -> bf16 convert --------------
__global__ __launch_bounds__(256) void cvt_bf16_kernel(
    const float* __restrict__ src, __bf16* __restrict__ dst, int n) {
  const int i = blockIdx.x * blockDim.x + threadIdx.x;
  if (i < n) dst[i] = f2bf(src[i]);
}

// --------- zero-padded copy of v_p (K-pad rows must be exact zeros) ---------
__global__ __launch_bounds__(256) void prep_vp_kernel(
    const __bf16* __restrict__ vb, __bf16* __restrict__ vpb) {
  const int r = blockIdx.x, h = blockIdx.y, d = threadIdx.x;   // r in [0,288)
  __bf16 val = (r < kMp) ? vb[((size_t)h * kNp + r) * kDHead + d] : f2bf(0.f);
  vpb[((size_t)h * kKpPad + r) * kDHead + d] = val;
}

// ---------------- WMMA GEMM ------------------------
// C[MxNb] = A[MxK] * B.  BTRANS: B stored row-major [Nb x K] (C = A*B^T),
//                        else:   B stored row-major [K  x Nb].
// All operands live in padded buffers -> NO load guards; K must be mult of 32
// with zero-filled K-pads.  Store guards (gm<M, gn<Nb) drop pad rows/cols.
// EPI 0: plain fp32 store.  EPI 1: QKV scatter epilogue.
template <bool BTRANS, int EPI>
__global__ __launch_bounds__(NTHR) void gemm_wmma_kernel(
    const __bf16* __restrict__ A, size_t strideAh, int lda,
    const __bf16* __restrict__ B, size_t strideBh, int ldb,
    float* __restrict__ C, size_t strideCh, int ldc,
    int M, int Nb, int K,
    __bf16* __restrict__ qo, __bf16* __restrict__ ko,
    __bf16* __restrict__ vo, float* __restrict__ vfo) {
  // A tile: [m][4 x u32x4]  (row = 32 bf16, pair-packed along K)
  // B tile: [n][4 x u32x4]  (col = 32 K-values, pair-packed along K)
  __shared__ u32x4 As[BM * 4];
  __shared__ u32x4 Bs[BN * 4];

  const int h = blockIdx.z;
  A += (size_t)h * strideAh;
  B += (size_t)h * strideBh;
  if (EPI == 0) C += (size_t)h * strideCh;

  const int m0 = blockIdx.x * BM, n0 = blockIdx.y * BN;
  const int tid = threadIdx.x, lane = tid & 31, wid = tid >> 5;
  const int wm = (wid >> 2) * 32, wn = (wid & 3) * 32;
  const int half = lane >> 4, l16 = lane & 15;

  // per-thread staging coordinates
  const int am = tid >> 2, asg = tid & 3;      // A: row, 8-elem segment
  const int bn = tid >> 1, bsg = tid & 1;      // B^T: col, 16-elem segment
  const int bkp = tid >> 4, bcs = tid & 15;    // B: k row-pair, 8-col segment

  const __bf16* aptr = A + (size_t)(m0 + am) * lda + asg * 8;
  const __bf16* bptr = BTRANS ? (B + (size_t)(n0 + bn) * ldb + bsg * 16)
                              : (B + (size_t)(2 * bkp) * ldb + n0 + bcs * 8);

  f32x8 acc00 = {}, acc01 = {}, acc10 = {}, acc11 = {};
  unsigned int* bsf = (unsigned int*)Bs;

  const int ksteps = K / BK;
  for (int kk = 0; kk < ksteps; ++kk) {
    // ---- stage A: one global b128 + one ds b128 per thread ----
    As[am * 4 + asg] = *(const u32x4*)aptr;
    // ---- stage B ----
    if (BTRANS) {
      const u32x4* s = (const u32x4*)bptr;
      Bs[bn * 4 + bsg * 2]     = s[0];
      Bs[bn * 4 + bsg * 2 + 1] = s[1];
    } else {
      u32x4 lo = *(const u32x4*)bptr;          // row 2*bkp,   8 cols
      u32x4 hi = *(const u32x4*)(bptr + ldb);  // row 2*bkp+1, 8 cols
      #pragma unroll
      for (int j = 0; j < 4; ++j) {            // pair-pack K into dwords
        bsf[(bcs * 8 + 2 * j    ) * 16 + bkp] = (lo[j] & 0xffffu) | (hi[j] << 16);
        bsf[(bcs * 8 + 2 * j + 1) * 16 + bkp] = (lo[j] >> 16) | (hi[j] & 0xffff0000u);
      }
    }
    aptr += BK;
    bptr += BTRANS ? (size_t)BK : (size_t)BK * ldb;
    if (kk + 1 < ksteps) {                     // global_prefetch next K-tile
      __builtin_prefetch(aptr, 0, 1);
      __builtin_prefetch(bptr, 0, 1);
    }
    __syncthreads();

    // ---- fragments: each = two ds_load_b128, k 8*half+0..7 then 16+8*half+0..7 ----
    const int ma0 = wm + l16, ma1 = wm + 16 + l16;
    const int nb0 = wn + l16, nb1 = wn + 16 + l16;
    bf16x16 a0 = cat16(As[ma0 * 4 + half], As[ma0 * 4 + 2 + half]);
    bf16x16 a1 = cat16(As[ma1 * 4 + half], As[ma1 * 4 + 2 + half]);
    bf16x16 b0 = cat16(Bs[nb0 * 4 + half], Bs[nb0 * 4 + 2 + half]);
    bf16x16 b1 = cat16(Bs[nb1 * 4 + half], Bs[nb1 * 4 + 2 + half]);

    acc00 = __builtin_amdgcn_wmma_f32_16x16x32_bf16(false, a0, false, b0, (short)0, acc00, false, false);
    acc01 = __builtin_amdgcn_wmma_f32_16x16x32_bf16(false, a0, false, b1, (short)0, acc01, false, false);
    acc10 = __builtin_amdgcn_wmma_f32_16x16x32_bf16(false, a1, false, b0, (short)0, acc10, false, false);
    acc11 = __builtin_amdgcn_wmma_f32_16x16x32_bf16(false, a1, false, b1, (short)0, acc11, false, false);

    __syncthreads();
  }

  // ---- epilogue: C/D layout = VGPR r -> M=r+8*half, N=lane&15 ----
  #pragma unroll
  for (int t = 0; t < 4; ++t) {
    const f32x8& acc = (t == 0) ? acc00 : (t == 1) ? acc01 : (t == 2) ? acc10 : acc11;
    const int mt = (t >> 1) * 16, nt = (t & 1) * 16;
    #pragma unroll
    for (int r = 0; r < 8; ++r) {
      const int gm = m0 + wm + mt + r + 8 * half;
      const int gn = n0 + wn + nt + l16;
      if (gm < M && gn < Nb) {
        const float c = acc[r];
        if (EPI == 0) {
          C[(size_t)gm * ldc + gn] = c;
        } else {  // QKV scatter: col -> (q|k|v, head, d)
          const int mat = gn >> 11;
          const int rem = gn & 2047;
          const int hh  = rem >> 8;
          const int d   = rem & 255;
          const size_t idx = ((size_t)hh * kNp + gm) * kDHead + d;
          if (mat == 0)      qo[idx] = f2bf(c * 0.0625f);   // dhead^-0.5
          else if (mat == 1) ko[idx] = f2bf(c);
          else               { vo[idx] = f2bf(c); vfo[idx] = c; }
        }
      }
    }
  }
}

// ---------------- row softmax (fp32 in, bf16 out, K-pad zeroed) ----
__global__ __launch_bounds__(256) void softmax_kernel(
    const float* __restrict__ S, size_t strideSh, int ldS,
    __bf16* __restrict__ P, size_t stridePh, int ldP, int cols) {
  const int row = blockIdx.x, h = blockIdx.y;
  const float* s = S + (size_t)h * strideSh + (size_t)row * ldS;
  __bf16* p = P + (size_t)h * stridePh + (size_t)row * ldP;
  const int tid = threadIdx.x, lane = tid & 31, w = tid >> 5;
  __shared__ float red[8];

  float mx = -3.0e38f;
  for (int j = tid; j < cols; j += 256) mx = fmaxf(mx, s[j]);
  #pragma unroll
  for (int o = 16; o; o >>= 1) mx = fmaxf(mx, __shfl_xor(mx, o, 32));
  if (!lane) red[w] = mx;
  __syncthreads();
  float m = red[0];
  #pragma unroll
  for (int i = 1; i < 8; ++i) m = fmaxf(m, red[i]);
  __syncthreads();

  float sum = 0.f;
  for (int j = tid; j < cols; j += 256) sum += __expf(s[j] - m);
  #pragma unroll
  for (int o = 16; o; o >>= 1) sum += __shfl_xor(sum, o, 32);
  if (!lane) red[w] = sum;
  __syncthreads();
  float ts = 0.f;
  #pragma unroll
  for (int i = 0; i < 8; ++i) ts += red[i];
  const float rinv = 1.f / ts;
  for (int j = tid; j < cols; j += 256) p[j] = f2bf(__expf(s[j] - m) * rinv);
  for (int j = cols + tid; j < ldP; j += 256) p[j] = f2bf(0.f);
}

// ---------------- attn-out + depthwise conv residual + interleave ----
__global__ __launch_bounds__(256) void combine_kernel(
    const float* __restrict__ attn,   // [h][n][d], stride kNp rows
    const float* __restrict__ vf,     // [h][n][d], stride kNp rows
    const float* __restrict__ rk,     // [h][33]
    float* __restrict__ out) {        // [n][h*256+d]
  const int n = blockIdx.x, h = blockIdx.y, d = threadIdx.x;
  const size_t base = (size_t)h * kNp * kDHead + d;
  float acc = attn[base + (size_t)n * kDHead];
  #pragma unroll
  for (int t = 0; t < kKRes; ++t) {
    const int nn = n + t - kPadC;
    if (nn >= 0 && nn < kN) acc += vf[base + (size_t)nn * kDHead] * rk[h * kKRes + t];
  }
  out[(size_t)n * kInner + h * kDHead + d] = acc;
}

// ---------------- host launcher --------------------
extern "C" void kernel_launch(void* const* d_in, const int* in_sizes, int n_in,
                              void* d_out, int out_size, void* d_ws, size_t ws_size,
                              hipStream_t stream) {
  (void)in_sizes; (void)n_in; (void)out_size; (void)ws_size;
  const float* x     = (const float*)d_in[0];
  const float* gamma = (const float*)d_in[1];
  const float* beta  = (const float*)d_in[2];
  const float* w     = (const float*)d_in[3];
  const float* rk    = (const float*)d_in[4];
  float* out = (float*)d_out;

  // workspace carve-out (256B aligned)
  char* ws = (char*)d_ws;
  size_t off = 0;
  auto alloc = [&](size_t bytes) -> char* {
    char* p = ws + off;
    off += (bytes + 255) & ~(size_t)255;
    return p;
  };
  const size_t hdsz = (size_t)kHeads * kNp * kDHead;     // padded per-head rows
  __bf16* xn  = (__bf16*)alloc((size_t)kNp * kDim * 2);
  __bf16* wb  = (__bf16*)alloc((size_t)kDim * kQKV * 2);
  __bf16* qb  = (__bf16*)alloc(hdsz * 2);
  __bf16* kb  = (__bf16*)alloc(hdsz * 2);
  __bf16* vb  = (__bf16*)alloc(hdsz * 2);
  float*  vf  = (float*) alloc(hdsz * 4);
  __bf16* vpb = (__bf16*)alloc((size_t)kHeads * kKpPad * kDHead * 2);  // v_p, K-padded w/ zeros
  __bf16* ph  = (__bf16*)alloc((size_t)kHeads * kNHist * kKpPad * 2);  // P_hist, K-pad zeroed
  __bf16* pp  = (__bf16*)alloc((size_t)kHeads * kMpPad * kNHist * 2);  // P_path, M-padded
  float* ows  = (float*) alloc(hdsz * 4);

  // output tuple offsets (flat, return order)
  float* o_ap  = out + (size_t)kN * kInner;
  float* o_cap = o_ap + (size_t)kHeads * kMp * kMp;
  float* o_cah = o_cap + (size_t)kHeads * kMp * kNHist;

  ln_kernel<<<kN, 256, 0, stream>>>(x, gamma, beta, xn);
  cvt_bf16_kernel<<<(kDim * kQKV + 255) / 256, 256, 0, stream>>>(w, wb, kDim * kQKV);

  const dim3 blk(NTHR);
  const size_t hs = (size_t)kNp * kDHead;

  // QKV GEMM with scatter epilogue: (8512x256)@(256x6144), stores guarded to 8476
  {
    dim3 g(kNp / BM, kQKV / BN, 1);
    gemm_wmma_kernel<false, 1><<<g, blk, 0, stream>>>(
        xn, 0, kDim, wb, 0, kQKV, nullptr, 0, 0,
        kN, kQKV, kDim, qb, kb, vb, vf);
  }
  prep_vp_kernel<<<dim3(kKpPad, kHeads), 256, 0, stream>>>(vb, vpb);

  // pre_softmax_cross_attn_histology = q_h @ k_p^T  (8192x284)
  {
    dim3 g(kNHist / BM, (kMp + BN - 1) / BN, kHeads);
    gemm_wmma_kernel<true, 0><<<g, blk, 0, stream>>>(
        qb + (size_t)kMp * kDHead, hs, kDHead, kb, hs, kDHead,
        o_cah, (size_t)kNHist * kMp, kMp,
        kNHist, kMp, kDHead, nullptr, nullptr, nullptr, nullptr);
  }
  // attn_pathways = q_p @ k_p^T  (284x284)
  {
    dim3 g(kMpPad / BM, (kMp + BN - 1) / BN, kHeads);
    gemm_wmma_kernel<true, 0><<<g, blk, 0, stream>>>(
        qb, hs, kDHead, kb, hs, kDHead,
        o_ap, (size_t)kMp * kMp, kMp,
        kMp, kMp, kDHead, nullptr, nullptr, nullptr, nullptr);
  }
  // cross_attn_pathways = q_p @ k_h^T  (284x8192)
  {
    dim3 g(kMpPad / BM, kNHist / BN, kHeads);
    gemm_wmma_kernel<true, 0><<<g, blk, 0, stream>>>(
        qb, hs, kDHead, kb + (size_t)kMp * kDHead, hs, kDHead,
        o_cap, (size_t)kMp * kNHist, kNHist,
        kMp, kNHist, kDHead, nullptr, nullptr, nullptr, nullptr);
  }
  // softmaxes -> bf16 probs
  softmax_kernel<<<dim3(kNHist, kHeads), 256, 0, stream>>>(
      o_cah, (size_t)kNHist * kMp, kMp, ph, (size_t)kNHist * kKpPad, kKpPad, kMp);
  softmax_kernel<<<dim3(kMp, kHeads), 256, 0, stream>>>(
      o_cap, (size_t)kMp * kNHist, kNHist, pp, (size_t)kMpPad * kNHist, kNHist, kNHist);
  // out_histology = P_hist @ v_p  (8192x256, K=288 zero-padded)
  {
    dim3 g(kNHist / BM, kDHead / BN, kHeads);
    gemm_wmma_kernel<false, 0><<<g, blk, 0, stream>>>(
        ph, (size_t)kNHist * kKpPad, kKpPad, vpb, (size_t)kKpPad * kDHead, kDHead,
        ows + (size_t)kMp * kDHead, hs, kDHead,
        kNHist, kDHead, kKpPad, nullptr, nullptr, nullptr, nullptr);
  }
  // out_pathways = P_path @ v_h  (284x256, K=8192)
  {
    dim3 g(kMpPad / BM, kDHead / BN, kHeads);
    gemm_wmma_kernel<false, 0><<<g, blk, 0, stream>>>(
        pp, (size_t)kMpPad * kNHist, kNHist, vb + (size_t)kMp * kDHead, hs, kDHead,
        ows, hs, kDHead,
        kMp, kDHead, kNHist, nullptr, nullptr, nullptr, nullptr);
  }
  // conv residual + concat + head interleave -> final out
  combine_kernel<<<dim3(kN, kHeads), 256, 0, stream>>>(ows, vf, rk, out);
}
